// BertSelfAttention_ling_35588099014686
// MI455X (gfx1250) — compile-verified, matrix operand order checked by reading, not run on hardware
//
#include <hip/hip_runtime.h>
#include <hip/hip_bf16.h>
#include <stdint.h>

// Problem constants
#define Bx   16
#define Sx   512
#define Hx   12
#define Dx   64
#define HIDx 768
#define NDIST 1023   // 2*MAX_POS-1

typedef __attribute__((ext_vector_type(16))) _Float16 v16h;
typedef __attribute__((ext_vector_type(8)))  _Float16 v8h;
typedef __attribute__((ext_vector_type(8)))  float    v8f;
typedef unsigned int u32x4 __attribute__((ext_vector_type(4)));
typedef int          i32x4 __attribute__((ext_vector_type(4)));
typedef int          i32x8 __attribute__((ext_vector_type(8)));

// Load 16 contiguous halves (16B-aligned) as one WMMA operand fragment's
// per-lane payload.  Fragment convention used consistently everywhere:
//   lane L (n=L&15, hi=L>>4):  row/col index = base + n,
//   K-halves = [chunk*32 + hi*16, +16).
// Since A and B of X @ Y^T are both loaded with the SAME K permutation,
// the 16x16x32 product is exact for any consistent packing.
__device__ __forceinline__ v16h load_frag16(const _Float16* p) {
  union { v16h v; v8h h[2]; } u;
  u.h[0] = *(const v8h*)(p);
  u.h[1] = *(const v8h*)(p + 8);
  return u.v;
}

__device__ __forceinline__ v8f wmma_f16(v16h a, v16h b, v8f c) {
  // (neg_a, A, neg_b, B, c_mod, C, reuse_a, reuse_b)
  return __builtin_amdgcn_wmma_f32_16x16x32_f16(false, a, false, b, (short)0, c,
                                                false, false);
}

// Per ISA 10.2 aperture calc, the low 32 bits of a flat shared address ARE
// the LDS byte offset.
__device__ __forceinline__ unsigned lds_off(const void* p) {
  return (unsigned)(size_t)p;
}

#if __has_builtin(__builtin_amdgcn_tensor_load_to_lds)
// TDM: DMA a 2D tile [rows x 64 halves] (row stride HIDx halves) into LDS.
// D# built per CDNA5 ISA ch.8 (group0 128b, group1 256b; groups 2/3 unused
// for 2D).  data_size=2B, tile_dim0=64, tile_dim1=rows, dim0_stride=HIDx.
__device__ __forceinline__ void tdm_load_tile_k(const _Float16* gsrc,
                                                unsigned ldsaddr, int rows) {
  unsigned long long ga = (unsigned long long)(size_t)gsrc;
  u32x4 g0;
  g0[0] = 1u;                                   // count=1, flags=0
  g0[1] = ldsaddr;                              // lds_addr (bytes)
  g0[2] = (unsigned)(ga & 0xffffffffu);         // global_addr[31:0]
  g0[3] = (unsigned)((ga >> 32) & 0x1ffffffu) | (2u << 30);  // addr[56:32]|type=2
  i32x8 g1;
  g1[0] = (int)(1u << 16);                      // data_size=1 -> 2 bytes
  g1[1] = (int)((64u & 0xffffu) << 16);         // tensor_dim0[15:0]
  g1[2] = (int)(((unsigned)rows & 0xffffu) << 16);  // td0 hi | tensor_dim1 lo
  g1[3] = (int)(64u << 16);                     // td1 hi | tile_dim0=64
  g1[4] = (int)((unsigned)rows & 0xffffu);      // tile_dim1=rows, tile_dim2=0
  g1[5] = (int)(unsigned)HIDx;                  // tensor_dim0_stride lo32
  g1[6] = 0;                                    // stride0 hi | stride1 lo
  g1[7] = 0;
  i32x4 gz = {0, 0, 0, 0};
#if __clang_major__ >= 23
  i32x8 gz8 = {0, 0, 0, 0, 0, 0, 0, 0};
  __builtin_amdgcn_tensor_load_to_lds(g0, g1, gz, gz, gz8, 0);
#else
  __builtin_amdgcn_tensor_load_to_lds(g0, g1, gz, gz, 0);
#endif
}
#endif

// ---------------------------------------------------------------- cvt ----
__global__ void cvt_f32_to_f16(const float* __restrict__ x,
                               _Float16* __restrict__ y, int n) {
  int i = blockIdx.x * blockDim.x + threadIdx.x;
  int stride = gridDim.x * blockDim.x;
  for (; i < n; i += stride) y[i] = (_Float16)x[i];
}

// ---------------------------------------------------------- QKV GEMM ----
// out[row, col] = sum_k hs[row,k] * W[col,k] + bias[col]   (X @ W^T + b)
// grid: (M/32, N/64, 3), block 256 (8 waves, each one 16x16 tile)
__global__ __launch_bounds__(256)
void qkv_gemm(const _Float16* __restrict__ hs,
              const _Float16* __restrict__ Wq, const _Float16* __restrict__ Wk,
              const _Float16* __restrict__ Wv,
              const float* __restrict__ bq, const float* __restrict__ bk,
              const float* __restrict__ bv,
              _Float16* __restrict__ Qo, _Float16* __restrict__ Ko,
              _Float16* __restrict__ Vo) {
  const int z = blockIdx.z;
  const _Float16* W    = (z == 0) ? Wq : (z == 1 ? Wk : Wv);
  const float*    bias = (z == 0) ? bq : (z == 1 ? bk : bv);
  _Float16*       out  = (z == 0) ? Qo : (z == 1 ? Ko : Vo);

  const int tid  = threadIdx.x;
  const int wave = tid >> 5, lane = tid & 31;
  const int n = lane & 15, hi = lane >> 4;
  const int mrow = blockIdx.x * 32 + (wave & 1) * 16;
  const int ncol = blockIdx.y * 64 + (wave >> 1) * 16;

  const _Float16* arow = hs + (size_t)(mrow + n) * HIDx + hi * 16;
  const _Float16* brow = W  + (size_t)(ncol + n) * HIDx + hi * 16;

  v8f acc = {};
#pragma unroll 4
  for (int k = 0; k < HIDx; k += 32) {
    v16h a = load_frag16(arow + k);
    v16h b = load_frag16(brow + k);
    acc = wmma_f16(a, b, acc);
  }

  const int col = ncol + n;
  const float bb = bias[col];
#pragma unroll
  for (int j = 0; j < 8; ++j) {
    const int row = mrow + j + 8 * hi;   // C layout: slot j -> M = j + 8*hi
    out[(size_t)row * HIDx + col] = (_Float16)(acc[j] + bb);
  }
}

// ------------------------------------------------- fused flash attention ----
// grid: (S/64, H, B), block 128 (4 waves; wave w owns l-rows [l0, l0+16))
__global__ __launch_bounds__(128)
void attn_fused(const _Float16* __restrict__ Q, const _Float16* __restrict__ K,
                const _Float16* __restrict__ V, const float* __restrict__ mask,
                const int* __restrict__ cmat, const int* __restrict__ wmat,
                const _Float16* __restrict__ dist,
                const float* __restrict__ cemb, const float* __restrict__ wemb,
                float* __restrict__ out) {
  __shared__ __align__(16) _Float16 KrS[128 * Dx];   // K chunk [rloc][d] (TDM)
  __shared__ __align__(16) _Float16 VtS[Dx * 128];   // V^T chunk [d][rloc]
  __shared__ __align__(16) _Float16 qdS[4][16 * 48]; // per-wave qd band
  __shared__ __align__(16) _Float16 kdS[4][32 * 48]; // per-wave kd band
  __shared__ __align__(16) _Float16 pS[4][16 * 32];  // per-wave prob tile
  __shared__ float ceS[8], weS[8];

  const int b = blockIdx.z, h = blockIdx.y;
  const int tid  = threadIdx.x;
  const int wave = tid >> 5, lane = tid & 31;
  const int n = lane & 15, hi = lane >> 4;
  const int l0 = blockIdx.x * 64 + wave * 16;

  if (tid < 8) { ceS[tid] = 0.5f * cemb[tid]; weS[tid] = 0.5f * wemb[tid]; }

  const _Float16* qbase = Q + (size_t)(b * Sx) * HIDx + h * Dx;
  const _Float16* kbase = K + (size_t)(b * Sx) * HIDx + h * Dx;
  const _Float16* vbase = V + (size_t)(b * Sx) * HIDx + h * Dx;

  // Persistent A fragments of q (K-chunks d=[0,32) and [32,64))
  const v16h qa0 = load_frag16(qbase + (size_t)(l0 + n) * HIDx + hi * 16);
  const v16h qa1 = load_frag16(qbase + (size_t)(l0 + n) * HIDx + 32 + hi * 16);

  v8f Oc0 = {}, Oc1 = {}, Oc2 = {}, Oc3 = {};
  float mrun[8], srun[8];
#pragma unroll
  for (int j = 0; j < 8; ++j) { mrun[j] = -1e30f; srun[j] = 0.f; }

  _Float16* qdw = qdS[wave];
  _Float16* kdw = kdS[wave];
  _Float16* pw  = pS[wave];

  for (int rsup = 0; rsup < 4; ++rsup) {
    __syncthreads();   // all waves done reading previous K/V chunk

#if __has_builtin(__builtin_amdgcn_tensor_load_to_lds)
    // Async DMA of the K chunk (TDM ignores EXEC; one issue from wave 0).
    if (wave == 0)
      tdm_load_tile_k(kbase + (size_t)(rsup * 128) * HIDx, lds_off(KrS), 128);
#else
    for (int t = tid; t < 1024; t += 128) {
      const int rloc = t >> 3, dg = (t & 7) * 8;
      *(v8h*)&KrS[rloc * Dx + dg] =
          *(const v8h*)(kbase + (size_t)(rsup * 128 + rloc) * HIDx + dg);
    }
#endif

    // Stage transposed V chunk with VALU while the TDM is in flight.
    for (int t = tid; t < 1024; t += 128) {
      const int rloc = t >> 3, dg = (t & 7) * 8;
      v8h vv = *(const v8h*)(vbase + (size_t)(rsup * 128 + rloc) * HIDx + dg);
#pragma unroll
      for (int i = 0; i < 8; ++i) VtS[(dg + i) * 128 + rloc] = vv[i];
    }

#if __has_builtin(__builtin_amdgcn_s_wait_tensorcnt) && \
    __has_builtin(__builtin_amdgcn_tensor_load_to_lds)
    if (wave == 0) __builtin_amdgcn_s_wait_tensorcnt(0);
#endif
    __syncthreads();   // publish K (TDM) + V (VALU) tiles

    for (int rc = 0; rc < 4; ++rc) {
      const int r0 = rsup * 128 + rc * 32;
      const int rl = rc * 32;                 // r offset inside LDS chunk
      // distance band: delta0 = l0 - r0 - 31; row index = delta + 511
      const int dzbase = l0 - r0 - 31 + 511;

      // pe B-fragments, band of 48 distances (3 N-tiles x 2 K-chunks)
      v16h pe[3][2];
#pragma unroll
      for (int tt = 0; tt < 3; ++tt) {
        int drow = dzbase + tt * 16 + n;
        drow = drow < 0 ? 0 : (drow > 1022 ? 1022 : drow);
        const _Float16* pr = dist + (size_t)drow * Dx;
        pe[tt][0] = load_frag16(pr + hi * 16);
        pe[tt][1] = load_frag16(pr + 32 + hi * 16);
      }

      // k fragments from the TDM-staged LDS chunk; used both as score
      // B-operands and as kd A-operands (dedup).
      const v16h kf00 = load_frag16(&KrS[(rl + n) * Dx + hi * 16]);
      const v16h kf01 = load_frag16(&KrS[(rl + n) * Dx + 32 + hi * 16]);
      const v16h kf10 = load_frag16(&KrS[(rl + 16 + n) * Dx + hi * 16]);
      const v16h kf11 = load_frag16(&KrS[(rl + 16 + n) * Dx + 32 + hi * 16]);

      // q @ k^T scores for r-subtiles [r0,r0+16) and [r0+16,r0+32)
      v8f c0 = {}, c1 = {};
      c0 = wmma_f16(qa0, kf00, c0);
      c0 = wmma_f16(qa1, kf01, c0);
      c1 = wmma_f16(qa0, kf10, c1);
      c1 = wmma_f16(qa1, kf11, c1);

      // qd[l, t] = q[l] . pe(delta0 + t)
#pragma unroll
      for (int tt = 0; tt < 3; ++tt) {
        v8f qd = {};
        qd = wmma_f16(qa0, pe[tt][0], qd);
        qd = wmma_f16(qa1, pe[tt][1], qd);
#pragma unroll
        for (int j = 0; j < 8; ++j)
          qdw[(j + 8 * hi) * 48 + tt * 16 + n] = (_Float16)qd[j];
      }

      // kd[r, t] = k[r] . pe(delta0 + t)
#pragma unroll
      for (int mt = 0; mt < 2; ++mt) {
        const v16h ka0 = mt ? kf10 : kf00;
        const v16h ka1 = mt ? kf11 : kf01;
#pragma unroll
        for (int tt = 0; tt < 3; ++tt) {
          v8f kd = {};
          kd = wmma_f16(ka0, pe[tt][0], kd);
          kd = wmma_f16(ka1, pe[tt][1], kd);
#pragma unroll
          for (int j = 0; j < 8; ++j)
            kdw[(mt * 16 + j + 8 * hi) * 48 + tt * 16 + n] = (_Float16)kd[j];
        }
      }

      const float mk0 = mask[b * Sx + r0 + n];
      const float mk1 = mask[b * Sx + r0 + 16 + n];
      const size_t cwbase = (size_t)b * Sx * Sx;

      // Prefetch next r-chunk's level-matrix rows into GL2 (bandwidth-
      // dominant stream): lane&15 picks the row, lane>>4 picks the matrix.
      if (r0 + 32 < Sx) {
        const size_t pf = cwbase + (size_t)(l0 + n) * Sx + (r0 + 32);
        __builtin_prefetch(hi ? (const void*)(wmat + pf)
                              : (const void*)(cmat + pf), 0, 1);
      }

      float e0[8], e1[8];
#pragma unroll
      for (int j = 0; j < 8; ++j) {
        const int ll = j + 8 * hi;
        const int l  = l0 + ll;
        const size_t rowb = cwbase + (size_t)l * Sx + r0;
        const int ci0 = cmat[rowb + n],      wi0 = wmat[rowb + n];
        const int ci1 = cmat[rowb + 16 + n], wi1 = wmat[rowb + 16 + n];
        const int t0 = ll - n + 31;          // in [16,46]
        const int t1 = ll - (16 + n) + 31;   // in [0,30]
        float s0 = (c0[j] + (float)qdw[ll * 48 + t0] + (float)kdw[n * 48 + t0]
                    + ceS[ci0] + weS[wi0]) * 0.125f + mk0;
        float s1 = (c1[j] + (float)qdw[ll * 48 + t1]
                    + (float)kdw[(16 + n) * 48 + t1]
                    + ceS[ci1] + weS[wi1]) * 0.125f + mk1;

        // online softmax: reduce across the 16 lanes holding this row's cols
        float vmax = fmaxf(s0, s1);
        vmax = fmaxf(vmax, __shfl_xor(vmax, 1, 32));
        vmax = fmaxf(vmax, __shfl_xor(vmax, 2, 32));
        vmax = fmaxf(vmax, __shfl_xor(vmax, 4, 32));
        vmax = fmaxf(vmax, __shfl_xor(vmax, 8, 32));
        const float mnew  = fmaxf(mrun[j], vmax);
        const float alpha = __expf(mrun[j] - mnew);
        mrun[j] = mnew;
        const float p0 = __expf(s0 - mnew);
        const float p1 = __expf(s1 - mnew);
        float rs = p0 + p1;
        rs += __shfl_xor(rs, 1, 32);
        rs += __shfl_xor(rs, 2, 32);
        rs += __shfl_xor(rs, 4, 32);
        rs += __shfl_xor(rs, 8, 32);
        srun[j] = srun[j] * alpha + rs;
        Oc0[j] *= alpha; Oc1[j] *= alpha; Oc2[j] *= alpha; Oc3[j] *= alpha;
        e0[j] = p0; e1[j] = p1;
      }

      // C-layout probs -> LDS -> A-fragment (transpose via LDS)
#pragma unroll
      for (int j = 0; j < 8; ++j) {
        const int ll = j + 8 * hi;
        pw[ll * 32 + n]      = (_Float16)e0[j];
        pw[ll * 32 + 16 + n] = (_Float16)e1[j];
      }
      v16h pa = load_frag16(pw + n * 32 + hi * 16);

      // O += p @ v   (B-fragments from transposed-V LDS)
      v16h vb0 = load_frag16(&VtS[(0 * 16 + n) * 128 + rl + hi * 16]);
      Oc0 = wmma_f16(pa, vb0, Oc0);
      v16h vb1 = load_frag16(&VtS[(1 * 16 + n) * 128 + rl + hi * 16]);
      Oc1 = wmma_f16(pa, vb1, Oc1);
      v16h vb2 = load_frag16(&VtS[(2 * 16 + n) * 128 + rl + hi * 16]);
      Oc2 = wmma_f16(pa, vb2, Oc2);
      v16h vb3 = load_frag16(&VtS[(3 * 16 + n) * 128 + rl + hi * 16]);
      Oc3 = wmma_f16(pa, vb3, Oc3);
    }
  }

  // epilogue: ctx[b, l, h*64 + d] = O / rowsum
#pragma unroll
  for (int j = 0; j < 8; ++j) {
    const int l = l0 + j + 8 * hi;
    const float inv = 1.0f / srun[j];
    float* orow = out + (size_t)(b * Sx + l) * HIDx + h * Dx + n;
    orow[0]  = Oc0[j] * inv;
    orow[16] = Oc1[j] * inv;
    orow[32] = Oc2[j] * inv;
    orow[48] = Oc3[j] * inv;
  }
}

// ----------------------------------------------------------- launcher ----
extern "C" void kernel_launch(void* const* d_in, const int* in_sizes, int n_in,
                              void* d_out, int out_size, void* d_ws,
                              size_t ws_size, hipStream_t stream) {
  (void)in_sizes; (void)n_in; (void)out_size; (void)ws_size;

  const float* hidden = (const float*)d_in[0];
  const float* mask   = (const float*)d_in[1];
  const int*   cmat   = (const int*)d_in[2];
  const int*   wmat   = (const int*)d_in[3];
  const float* Wq     = (const float*)d_in[4];
  const float* bq     = (const float*)d_in[5];
  const float* Wk     = (const float*)d_in[6];
  const float* bk     = (const float*)d_in[7];
  const float* Wv     = (const float*)d_in[8];
  const float* bv     = (const float*)d_in[9];
  const float* dist   = (const float*)d_in[10];
  const float* cemb   = (const float*)d_in[11];
  const float* wemb   = (const float*)d_in[12];
  float* out = (float*)d_out;

  // f16 workspace layout (~54 MB total)
  _Float16* ws = (_Float16*)d_ws;
  const size_t N_HS = (size_t)Bx * Sx * HIDx;   // 6,291,456
  const size_t N_W  = (size_t)HIDx * HIDx;      //   589,824
  const size_t N_D  = (size_t)NDIST * Dx;       //    65,472
  _Float16* hs16 = ws;
  _Float16* Wq16 = hs16 + N_HS;
  _Float16* Wk16 = Wq16 + N_W;
  _Float16* Wv16 = Wk16 + N_W;
  _Float16* d16  = Wv16 + N_W;
  _Float16* Q16  = d16 + N_D;
  _Float16* K16  = Q16 + N_HS;
  _Float16* V16  = K16 + N_HS;

  auto cvt = [&](const float* src, _Float16* dst, size_t n) {
    int blocks = (int)((n + 255) / 256);
    if (blocks > 4096) blocks = 4096;
    cvt_f32_to_f16<<<blocks, 256, 0, stream>>>(src, dst, (int)n);
  };
  cvt(hidden, hs16, N_HS);
  cvt(Wq, Wq16, N_W);
  cvt(Wk, Wk16, N_W);
  cvt(Wv, Wv16, N_W);
  cvt(dist, d16, N_D);

  qkv_gemm<<<dim3((Bx * Sx) / 32, HIDx / 64, 3), 256, 0, stream>>>(
      hs16, Wq16, Wk16, Wv16, bq, bk, bv, Q16, K16, V16);

  attn_fused<<<dim3(Sx / 64, Hx, Bx), 128, 0, stream>>>(
      Q16, K16, V16, mask, cmat, wmat, d16, cemb, wemb, out);
}
